// DGCNN_jigsaw_15771119911615
// MI455X (gfx1250) — matrix operand order repaired and checked by reading, not compiled, not run on hardware
//
#include <hip/hip_runtime.h>
#include <hip/hip_bf16.h>

// ---------------------------------------------------------------------------
// DGCNN forward for MI455X (gfx1250, wave32, WMMA).
//
//   transpose -> [knn(WMMA gram) -> edgeconv(WMMA)+stats -> finalize ->
//   conv(WMMA, async-LDS staged)+stats -> finalize -> max_k] x2
//   -> [knn -> edgeconv -> finalize -> max_k]
//   -> final GEMM(K=192,N=1024, WMMA)+stats -> finalize -> bn+transpose+argmax
//
// f16 feature/weight storage: every WMMA fragment is two b128 loads (global or
// LDS), no per-element f32->f16 conversion in the hot loops.
// ---------------------------------------------------------------------------

#define DEV __device__ __forceinline__

typedef __attribute__((ext_vector_type(16))) _Float16 v16h;
typedef __attribute__((ext_vector_type(8)))  _Float16 h8;
typedef __attribute__((ext_vector_type(8)))  float    v8f;
typedef int v4i_vs __attribute__((__vector_size__(4 * sizeof(int))));

static constexpr int BB  = 16;
static constexpr int NN  = 2048;
static constexpr int KNN = 20;
static constexpr float NEG = -3.402823466e38f;

#if defined(__has_builtin)
#if __has_builtin(__builtin_amdgcn_global_load_async_to_lds_b128) && \
    __has_builtin(__builtin_amdgcn_s_wait_asynccnt)
#define HAVE_ASYNC_LDS 1
#endif
#endif
#ifndef HAVE_ASYNC_LDS
#define HAVE_ASYNC_LDS 0
#endif

DEV v8f wmma16(v16h a, v16h b, v8f c) {
  return __builtin_amdgcn_wmma_f32_16x16x32_f16(false, a, false, b,
                                                (short)0, c, false, false);
}
// A fragment K map: K = e + 8*(e>=8) + 8*(lane>=16)  => two contiguous 8-half
// chunks at k0 = kbase + 8*(lane>=16) and k0+16.
DEV int a_kmap(int e, int lane) { return e + ((e >= 8) ? 8 : 0) + ((lane >= 16) ? 8 : 0); }
DEV v16h ld_afrag(const _Float16* row, int lane, int kbase) {
  const int k0 = kbase + ((lane >> 4) << 3);
  h8 lo = *(const h8*)(row + k0);
  h8 hi = *(const h8*)(row + k0 + 16);
  return __builtin_shufflevector(lo, hi, 0, 1, 2, 3, 4, 5, 6, 7,
                                 8, 9, 10, 11, 12, 13, 14, 15);
}
// B fragment K map: same as A: K = e + 8*(e>=8) + 8*(lane>=16).
DEV v16h ld_bfrag(const _Float16* row, int lane, int kbase) {
  const int k0 = kbase + ((lane >> 4) << 3);
  h8 lo = *(const h8*)(row + k0);
  h8 hi = *(const h8*)(row + k0 + 16);
  return __builtin_shufflevector(lo, hi, 0, 1, 2, 3, 4, 5, 6, 7,
                                 8, 9, 10, 11, 12, 13, 14, 15);
}

DEV float lrelu(float v) { return v > 0.f ? v : 0.2f * v; }

DEV v16h bn_lrelu16(v16h v, v16h s, v16h t) {
  v16h r;
#pragma unroll
  for (int e = 0; e < 16; ++e) {
    _Float16 x = v[e] * s[e] + t[e];
    r[e] = (x > (_Float16)0) ? x : (_Float16)0.2f * x;
  }
  return r;
}

// ---------------------------------------------------------------------------
// kNN, C==64 (f16 features): WMMA gram tile -> 16x2048 scores in LDS ->
// iterative top-20 per row with wave32 shfl reductions.
// ---------------------------------------------------------------------------
__global__ void knn64_kernel(const _Float16* __restrict__ X,
                             const float* __restrict__ SQ,
                             int* __restrict__ OUT) {
  extern __shared__ float s_d[];  // [16][NN]
  const int b    = blockIdx.x / (NN / 16);
  const int r0   = (blockIdx.x % (NN / 16)) * 16;
  const int tid  = threadIdx.x;
  const int lane = tid & 31;
  const int wave = tid >> 5;

  const int m = lane & 15;
  const _Float16* xr = X + (size_t)(b * NN + r0 + m) * 64;
  const v16h a0 = ld_afrag(xr, lane, 0);
  const v16h a1 = ld_afrag(xr, lane, 32);
  float sqr[8];
#pragma unroll
  for (int v = 0; v < 8; ++v)
    sqr[v] = SQ[b * NN + r0 + v + ((lane >> 4) << 3)];

  for (int t = wave; t < NN / 16; t += 8) {
    const int c0 = t * 16;
    const _Float16* xc = X + (size_t)(b * NN + c0 + m) * 64;
    __builtin_prefetch(X + (size_t)(b * NN + c0 + 128 + m) * 64, 0, 1);
    v8f acc = {};
    acc = wmma16(a0, ld_bfrag(xc, lane, 0), acc);
    acc = wmma16(a1, ld_bfrag(xc, lane, 32), acc);
    const float sqc = SQ[b * NN + c0 + m];
#pragma unroll
    for (int v = 0; v < 8; ++v) {
      const int rr = v + ((lane >> 4) << 3);
      s_d[rr * NN + c0 + m] = 2.0f * acc[v] - sqr[v] - sqc;
    }
  }
  __syncthreads();

  for (int it = 0; it < KNN; ++it) {
#pragma unroll
    for (int rr = 0; rr < 2; ++rr) {
      const int r = wave + rr * 8;
      float best = NEG;
      int bi = 0;
      for (int c = lane; c < NN; c += 32) {
        const float v = s_d[r * NN + c];
        if (v > best) { best = v; bi = c; }
      }
#pragma unroll
      for (int off = 16; off > 0; off >>= 1) {
        const float ov = __shfl_xor(best, off, 32);
        const int   oi = __shfl_xor(bi, off, 32);
        if (ov > best || (ov == best && oi < bi)) { best = ov; bi = oi; }
      }
      if (lane == 0) {
        OUT[(size_t)(b * NN + r0 + r) * KNN + it] = bi;
        s_d[r * NN + bi] = NEG;
      }
    }
    __syncthreads();
  }
}

// kNN for xyz (f32, padded stride 4): VALU path (K=3 would waste the WMMA pipe)
__global__ void knn4_kernel(const float* __restrict__ X,
                            const float* __restrict__ SQ,
                            int* __restrict__ OUT) {
  extern __shared__ float s_d[];  // [16][NN] + 64
  const int b    = blockIdx.x / (NN / 16);
  const int r0   = (blockIdx.x % (NN / 16)) * 16;
  const int tid  = threadIdx.x;
  const int lane = tid & 31;
  const int wave = tid >> 5;

  float* s_rows = s_d + 16 * NN;
  if (tid < 64) {
    const int r = tid >> 2, c = tid & 3;
    s_rows[tid] = X[(size_t)(b * NN + r0 + r) * 4 + c];
  }
  __syncthreads();
  float sqr[16];
#pragma unroll
  for (int r = 0; r < 16; ++r) sqr[r] = SQ[b * NN + r0 + r];
  for (int col = tid; col < NN; col += 256) {
    const float cx = X[(size_t)(b * NN + col) * 4 + 0];
    const float cy = X[(size_t)(b * NN + col) * 4 + 1];
    const float cz = X[(size_t)(b * NN + col) * 4 + 2];
    const float sqc = SQ[b * NN + col];
#pragma unroll
    for (int r = 0; r < 16; ++r) {
      const float d = cx * s_rows[r * 4] + cy * s_rows[r * 4 + 1] +
                      cz * s_rows[r * 4 + 2];
      s_d[r * NN + col] = 2.0f * d - sqr[r] - sqc;
    }
  }
  __syncthreads();

  for (int it = 0; it < KNN; ++it) {
#pragma unroll
    for (int rr = 0; rr < 2; ++rr) {
      const int r = wave + rr * 8;
      float best = NEG;
      int bi = 0;
      for (int c = lane; c < NN; c += 32) {
        const float v = s_d[r * NN + c];
        if (v > best) { best = v; bi = c; }
      }
#pragma unroll
      for (int off = 16; off > 0; off >>= 1) {
        const float ov = __shfl_xor(best, off, 32);
        const int   oi = __shfl_xor(bi, off, 32);
        if (ov > best || (ov == best && oi < bi)) { best = ov; bi = oi; }
      }
      if (lane == 0) {
        OUT[(size_t)(b * NN + r0 + r) * KNN + it] = bi;
        s_d[r * NN + bi] = NEG;
      }
    }
    __syncthreads();
  }
}

__global__ void rowsq4_kernel(const float* __restrict__ X, float* __restrict__ SQ) {
  const int i = blockIdx.x * 256 + threadIdx.x;  // B*N
  const float* p = X + (size_t)i * 4;
  SQ[i] = p[0] * p[0] + p[1] * p[1] + p[2] * p[2];
}

__global__ void rowsq64h_kernel(const _Float16* __restrict__ X, float* __restrict__ SQ) {
  const int i = blockIdx.x * 256 + threadIdx.x;  // B*N
  const _Float16* p = X + (size_t)i * 64;
  float s = 0.f;
#pragma unroll 16
  for (int c = 0; c < 64; ++c) { const float v = (float)p[c]; s += v * v; }
  SQ[i] = s;
}

// (B,3,N) channels-first -> (B,N,4) row-major, pad[3]=0
__global__ void transpose_kernel(const float* __restrict__ X, float* __restrict__ XT) {
  const int i = blockIdx.x * 256 + threadIdx.x;  // B*N
  const int b = i >> 11, n = i & (NN - 1);
  XT[(size_t)i * 4 + 0] = X[((size_t)b * 3 + 0) * NN + n];
  XT[(size_t)i * 4 + 1] = X[((size_t)b * 3 + 1) * NN + n];
  XT[(size_t)i * 4 + 2] = X[((size_t)b * 3 + 2) * NN + n];
  XT[(size_t)i * 4 + 3] = 0.f;
}

__global__ void zero_kernel(float* __restrict__ p, int n) {
  const int i = blockIdx.x * 256 + threadIdx.x;
  if (i < n) p[i] = 0.f;
}

// f32 weights -> zero-padded f16 rows (row length kpad)
__global__ void convw_kernel(const float* __restrict__ W, _Float16* __restrict__ Wh,
                             int rows, int cols, int kpad) {
  const int i = blockIdx.x * 256 + threadIdx.x;
  if (i >= rows * kpad) return;
  const int r = i / kpad, c = i % kpad;
  Wh[i] = (_Float16)((c < cols) ? W[r * cols + c] : 0.f);
}

// ---------------------------------------------------------------------------
// Edge conv stage 1: f = [x[nbr]-x[n], x[n]] (K=6 padded to 32) x W1h^T.
// ---------------------------------------------------------------------------
__global__ void conv_edge3_kernel(const float* __restrict__ XT,
                                  const int* __restrict__ IDX,
                                  const _Float16* __restrict__ W1h,  // 64x32
                                  _Float16* __restrict__ Y,
                                  float* __restrict__ stats) {
  __shared__ float s_stat[128];
  const int tid = threadIdx.x, lane = tid & 31, wave = tid >> 5;
  if (tid < 128) s_stat[tid] = 0.f;
  __syncthreads();

  const int m0   = blockIdx.x * 32 + (wave >> 2) * 16;
  const int mrow = m0 + (lane & 15);
  const int b    = mrow / (NN * KNN);
  const int rem  = mrow % (NN * KNN);
  const int n    = rem / KNN;
  const int kk   = rem % KNN;
  const int nbr  = IDX[(size_t)(b * NN + n) * KNN + kk];
  const float* xc = XT + (size_t)(b * NN + n) * 4;
  const float* xn = XT + (size_t)(b * NN + nbr) * 4;

  v16h af{};
#pragma unroll
  for (int e = 0; e < 16; ++e) {
    const int k = a_kmap(e, lane);
    float v = 0.f;
    if (k < 3)      v = xn[k] - xc[k];
    else if (k < 6) v = xc[k - 3];
    af[e] = (_Float16)v;
  }
  const int o = (wave & 3) * 16 + (lane & 15);
  v8f acc = {};
  acc = wmma16(af, ld_bfrag(W1h + o * 32, lane, 0), acc);

  float lsum = 0.f, lsq = 0.f;
#pragma unroll
  for (int v = 0; v < 8; ++v) {
    const int row = m0 + v + ((lane >> 4) << 3);
    const float val = acc[v];
    Y[(size_t)row * 64 + o] = (_Float16)val;
    lsum += val;
    lsq  += val * val;
  }
  atomicAdd(&s_stat[o], lsum);
  atomicAdd(&s_stat[64 + o], lsq);
  __syncthreads();
  if (tid < 128) atomicAdd(&stats[tid], s_stat[tid]);
}

// ---------------------------------------------------------------------------
// Edge conv stages 2/3: K=128 (diff | center), all-f16 fragment math.
// ---------------------------------------------------------------------------
__global__ void conv_edge64_kernel(const _Float16* __restrict__ XIN,
                                   const int* __restrict__ IDX,
                                   const _Float16* __restrict__ Wh,  // 64x128
                                   _Float16* __restrict__ Y,
                                   float* __restrict__ stats) {
  __shared__ float s_stat[128];
  const int tid = threadIdx.x, lane = tid & 31, wave = tid >> 5;
  if (tid < 128) s_stat[tid] = 0.f;
  __syncthreads();

  const int m0   = blockIdx.x * 32 + (wave >> 2) * 16;
  const int mrow = m0 + (lane & 15);
  const int b    = mrow / (NN * KNN);
  const int rem  = mrow % (NN * KNN);
  const int n    = rem / KNN;
  const int kk   = rem % KNN;
  const int nbr  = IDX[(size_t)(b * NN + n) * KNN + kk];
  const _Float16* xc = XIN + (size_t)(b * NN + n) * 64;
  const _Float16* xn = XIN + (size_t)(b * NN + nbr) * 64;

  v16h af[4];
  af[0] = ld_afrag(xn, lane, 0)  - ld_afrag(xc, lane, 0);
  af[1] = ld_afrag(xn, lane, 32) - ld_afrag(xc, lane, 32);
  af[2] = ld_afrag(xc, lane, 0);
  af[3] = ld_afrag(xc, lane, 32);

  const int o = (wave & 3) * 16 + (lane & 15);
  const _Float16* wr = Wh + o * 128;
  v8f acc = {};
#pragma unroll
  for (int ks = 0; ks < 4; ++ks)
    acc = wmma16(af[ks], ld_bfrag(wr, lane, ks * 32), acc);

  float lsum = 0.f, lsq = 0.f;
#pragma unroll
  for (int v = 0; v < 8; ++v) {
    const int row = m0 + v + ((lane >> 4) << 3);
    const float val = acc[v];
    Y[(size_t)row * 64 + o] = (_Float16)val;
    lsum += val;
    lsq  += val * val;
  }
  atomicAdd(&s_stat[o], lsum);
  atomicAdd(&s_stat[64 + o], lsq);
  __syncthreads();
  if (tid < 128) atomicAdd(&stats[tid], s_stat[tid]);
}

// ---------------------------------------------------------------------------
// Second conv of a stage: A = lrelu(bn(Yprev)) (K=64).  A tile staged into
// LDS with CDNA5 async load-to-LDS (fallback: sync b128 copy), fragments read
// back as ds_load_b128 (rows padded to 72 halves -> conflict-free).
// ---------------------------------------------------------------------------
__global__ void conv_plain_kernel(const _Float16* __restrict__ YIN,
                                  const float* __restrict__ SS,   // [scale64,shift64]
                                  const _Float16* __restrict__ Wh,  // 64x64
                                  _Float16* __restrict__ Y,
                                  float* __restrict__ stats) {
  __shared__ float s_stat[128];
  __shared__ _Float16 s_a[32 * 72];
  const int tid = threadIdx.x, lane = tid & 31, wave = tid >> 5;
  if (tid < 128) s_stat[tid] = 0.f;

  const int mblk = blockIdx.x * 32;
  {
    const int row = tid >> 3, ch = (tid & 7) * 8;
    const _Float16* src = YIN + (size_t)(mblk + row) * 64 + ch;
    _Float16* dst = s_a + row * 72 + ch;
#if HAVE_ASYNC_LDS
    __builtin_amdgcn_global_load_async_to_lds_b128(
        (__attribute__((address_space(1))) v4i_vs*)src,
        (__attribute__((address_space(3))) v4i_vs*)dst, 0, 0);
    __builtin_amdgcn_s_wait_asynccnt(0);
#else
    *(h8*)dst = *(const h8*)src;
#endif
  }
  __syncthreads();

  // per-lane bn scale/shift fragments (f16)
  v16h sf[2], tf[2];
#pragma unroll
  for (int ks = 0; ks < 2; ++ks)
#pragma unroll
    for (int e = 0; e < 16; ++e) {
      const int k = ks * 32 + a_kmap(e, lane);
      sf[ks][e] = (_Float16)SS[k];
      tf[ks][e] = (_Float16)SS[64 + k];
    }

  const _Float16* yr = s_a + ((wave >> 2) * 16 + (lane & 15)) * 72;
  v16h af[2];
  af[0] = bn_lrelu16(ld_afrag(yr, lane, 0),  sf[0], tf[0]);
  af[1] = bn_lrelu16(ld_afrag(yr, lane, 32), sf[1], tf[1]);

  const int m0 = mblk + (wave >> 2) * 16;
  const int o  = (wave & 3) * 16 + (lane & 15);
  const _Float16* wr = Wh + o * 64;
  v8f acc = {};
  acc = wmma16(af[0], ld_bfrag(wr, lane, 0),  acc);
  acc = wmma16(af[1], ld_bfrag(wr, lane, 32), acc);

  float lsum = 0.f, lsq = 0.f;
#pragma unroll
  for (int v = 0; v < 8; ++v) {
    const int row = m0 + v + ((lane >> 4) << 3);
    const float val = acc[v];
    Y[(size_t)row * 64 + o] = (_Float16)val;
    lsum += val;
    lsq  += val * val;
  }
  atomicAdd(&s_stat[o], lsum);
  atomicAdd(&s_stat[64 + o], lsq);
  __syncthreads();
  if (tid < 128) atomicAdd(&stats[tid], s_stat[tid]);
}

// stats -> (scale, shift):  s = g*rsqrt(var+eps), t = b - mean*s
__global__ void finalize_stats_kernel(const float* __restrict__ stats,
                                      const float* __restrict__ g,
                                      const float* __restrict__ bta,
                                      float* __restrict__ SS, int C, float inv) {
  const int c = blockIdx.x * 256 + threadIdx.x;
  if (c >= C) return;
  const float mean = stats[c] * inv;
  const float var  = stats[C + c] * inv - mean * mean;
  const float s    = g[c] * rsqrtf(var + 1e-5f);
  SS[c]     = s;
  SS[C + c] = bta[c] - mean * s;
}

// x_out[b,n,c] = max_k lrelu(bn(y[b,n,k,c]))  (f16 out)
__global__ void maxk_kernel(const _Float16* __restrict__ YIN,
                            const float* __restrict__ SS,
                            _Float16* __restrict__ XOUT) {
  const int t  = blockIdx.x * 256 + threadIdx.x;  // B*N*64
  const int c  = t & 63;
  const int mn = t >> 6;
  const float s = SS[c], sh = SS[64 + c];
  float best = NEG;
#pragma unroll
  for (int kk = 0; kk < KNN; ++kk) {
    const float v = lrelu((float)YIN[((size_t)mn * KNN + kk) * 64 + c] * s + sh);
    best = fmaxf(best, v);
  }
  XOUT[(size_t)mn * 64 + c] = (_Float16)best;
}

// ---------------------------------------------------------------------------
// Final GEMM: cat(x1,x2,x3) (M=B*N, K=192, f16) x W6h^T (1024).
// ---------------------------------------------------------------------------
__global__ void final_gemm_kernel(const _Float16* __restrict__ X1,
                                  const _Float16* __restrict__ X2,
                                  const _Float16* __restrict__ X3,
                                  const _Float16* __restrict__ W6h,  // 1024x192
                                  _Float16* __restrict__ YP,
                                  float* __restrict__ stats) {
  __shared__ float s_stat[2048];
  const int tid = threadIdx.x, lane = tid & 31, wave = tid >> 5;
  for (int i = tid; i < 2048; i += 256) s_stat[i] = 0.f;
  __syncthreads();

  const int m0   = blockIdx.x * 32 + (wave >> 2) * 16;
  const int mrow = m0 + (lane & 15);
  const _Float16* s0 = X1 + (size_t)mrow * 64;
  const _Float16* s1 = X2 + (size_t)mrow * 64;
  const _Float16* s2 = X3 + (size_t)mrow * 64;
  v16h a[6];
  a[0] = ld_afrag(s0, lane, 0);  a[1] = ld_afrag(s0, lane, 32);
  a[2] = ld_afrag(s1, lane, 0);  a[3] = ld_afrag(s1, lane, 32);
  a[4] = ld_afrag(s2, lane, 0);  a[5] = ld_afrag(s2, lane, 32);

  for (int ct = (wave & 3); ct < 64; ct += 4) {
    const int o = ct * 16 + (lane & 15);
    const _Float16* wr = W6h + (size_t)o * 192;
    v8f acc = {};
#pragma unroll
    for (int ks = 0; ks < 6; ++ks)
      acc = wmma16(a[ks], ld_bfrag(wr, lane, ks * 32), acc);

    float lsum = 0.f, lsq = 0.f;
#pragma unroll
    for (int v = 0; v < 8; ++v) {
      const int row = m0 + v + ((lane >> 4) << 3);
      const float val = acc[v];
      YP[(size_t)row * 1024 + o] = (_Float16)val;
      lsum += val;
      lsq  += val * val;
    }
    atomicAdd(&s_stat[o], lsum);
    atomicAdd(&s_stat[1024 + o], lsq);
  }
  __syncthreads();
  for (int i = tid; i < 2048; i += 256) atomicAdd(&stats[i], s_stat[i]);
}

// bn+lrelu, transpose (B,N,1024)->(B,1024,N) via LDS tile, fused argmax/max
__global__ void final_out_kernel(const _Float16* __restrict__ YP,
                                 const float* __restrict__ SS,
                                 float* __restrict__ YOUT,
                                 int* __restrict__ CIDX,
                                 float* __restrict__ POOL) {
  __shared__ float tile[64 * 65];
  const int b  = blockIdx.x >> 4;
  const int o0 = (blockIdx.x & 15) * 64;
  const int t  = threadIdx.x;  // 64 threads
  const int o  = o0 + t;
  const float s = SS[o], sh = SS[1024 + o];
  float best = NEG;
  int bidx = 0;
  for (int n0 = 0; n0 < NN; n0 += 64) {
    for (int nn = 0; nn < 64; ++nn) {
      float v = (float)YP[((size_t)(b * NN + n0 + nn)) * 1024 + o];
      v = lrelu(v * s + sh);
      tile[t * 65 + nn] = v;
      if (v > best) { best = v; bidx = n0 + nn; }
    }
    __syncthreads();
    for (int oo = 0; oo < 64; ++oo)
      YOUT[((size_t)b * 1024 + o0 + oo) * NN + n0 + t] = tile[oo * 65 + t];
    __syncthreads();
  }
  CIDX[b * 1024 + o] = bidx;
  POOL[b * 1024 + o] = best;
}

// ---------------------------------------------------------------------------
extern "C" void kernel_launch(void* const* d_in, const int* in_sizes, int n_in,
                              void* d_out, int out_size, void* d_ws, size_t ws_size,
                              hipStream_t stream) {
  (void)in_sizes; (void)n_in; (void)out_size; (void)ws_size;
  const float* x  = (const float*)d_in[0];
  const float* W1 = (const float*)d_in[1];
  const float* W2 = (const float*)d_in[2];
  const float* W3 = (const float*)d_in[3];
  const float* W4 = (const float*)d_in[4];
  const float* W5 = (const float*)d_in[5];
  const float* W6 = (const float*)d_in[6];
  const float* g[6], *bt[6];
  for (int i = 0; i < 6; ++i) { g[i] = (const float*)d_in[7 + 2 * i]; bt[i] = (const float*)d_in[8 + 2 * i]; }

  char* ws = (char*)d_ws;
  size_t off = 0;
  auto carve = [&](size_t bytes) { char* p = ws + off; off += (bytes + 255) & ~(size_t)255; return p; };
  float*     xt   = (float*)carve((size_t)BB * NN * 4 * 4);
  float*     sq   = (float*)carve((size_t)BB * NN * 4);
  int*       idx  = (int*)carve((size_t)BB * NN * KNN * 4);
  _Float16*  x1   = (_Float16*)carve((size_t)BB * NN * 64 * 2);
  _Float16*  x2   = (_Float16*)carve((size_t)BB * NN * 64 * 2);
  _Float16*  x3   = (_Float16*)carve((size_t)BB * NN * 64 * 2);
  _Float16*  y1h  = (_Float16*)carve((size_t)BB * NN * KNN * 64 * 2);
  _Float16*  y2h  = (_Float16*)carve((size_t)BB * NN * KNN * 64 * 2);
  _Float16*  yph  = (_Float16*)carve((size_t)BB * NN * 1024 * 2);
  _Float16*  w1h  = (_Float16*)carve(64 * 32 * 2);
  _Float16*  w2h  = (_Float16*)carve(64 * 64 * 2);
  _Float16*  w3h  = (_Float16*)carve(64 * 128 * 2);
  _Float16*  w4h  = (_Float16*)carve(64 * 64 * 2);
  _Float16*  w5h  = (_Float16*)carve(64 * 128 * 2);
  _Float16*  w6h  = (_Float16*)carve((size_t)1024 * 192 * 2);
  float*     stats = (float*)carve(2688 * 4);  // bn1..bn5: 128 ea, bn6: 2048
  float*     ssb   = (float*)carve(2688 * 4);
  float* st[6] = {stats, stats + 128, stats + 256, stats + 384, stats + 512, stats + 640};
  float* ss[6] = {ssb,   ssb + 128,   ssb + 256,   ssb + 384,   ssb + 512,   ssb + 640};

  const int BN = BB * NN;   // 32768
  const int Mk = BN * KNN;  // 655360 edge rows
  const size_t knn_lds = (size_t)(16 * NN + 64) * 4;
  const float invK  = 1.f / (float)Mk;
  const float invBN = 1.f / (float)BN;

  zero_kernel<<<(2688 + 255) / 256, 256, 0, stream>>>(stats, 2688);
  convw_kernel<<<(64 * 32 + 255) / 256, 256, 0, stream>>>(W1, w1h, 64, 6, 32);
  convw_kernel<<<(64 * 64 + 255) / 256, 256, 0, stream>>>(W2, w2h, 64, 64, 64);
  convw_kernel<<<(64 * 128 + 255) / 256, 256, 0, stream>>>(W3, w3h, 64, 128, 128);
  convw_kernel<<<(64 * 64 + 255) / 256, 256, 0, stream>>>(W4, w4h, 64, 64, 64);
  convw_kernel<<<(64 * 128 + 255) / 256, 256, 0, stream>>>(W5, w5h, 64, 128, 128);
  convw_kernel<<<(1024 * 192 + 255) / 256, 256, 0, stream>>>(W6, w6h, 1024, 192, 192);

  transpose_kernel<<<BN / 256, 256, 0, stream>>>(x, xt);
  rowsq4_kernel<<<BN / 256, 256, 0, stream>>>(xt, sq);

  // ---- stage 1 ----
  knn4_kernel<<<BB * (NN / 16), 256, knn_lds, stream>>>(xt, sq, idx);
  conv_edge3_kernel<<<Mk / 32, 256, 0, stream>>>(xt, idx, w1h, y1h, st[0]);
  finalize_stats_kernel<<<1, 256, 0, stream>>>(st[0], g[0], bt[0], ss[0], 64, invK);
  conv_plain_kernel<<<Mk / 32, 256, 0, stream>>>(y1h, ss[0], w2h, y2h, st[1]);
  finalize_stats_kernel<<<1, 256, 0, stream>>>(st[1], g[1], bt[1], ss[1], 64, invK);
  maxk_kernel<<<(BN * 64) / 256, 256, 0, stream>>>(y2h, ss[1], x1);

  // ---- stage 2 ----
  rowsq64h_kernel<<<BN / 256, 256, 0, stream>>>(x1, sq);
  knn64_kernel<<<BB * (NN / 16), 256, knn_lds, stream>>>(x1, sq, idx);
  conv_edge64_kernel<<<Mk / 32, 256, 0, stream>>>(x1, idx, w3h, y1h, st[2]);
  finalize_stats_kernel<<<1, 256, 0, stream>>>(st[2], g[2], bt[2], ss[2], 64, invK);
  conv_plain_kernel<<<Mk / 32, 256, 0, stream>>>(y1h, ss[2], w4h, y2h, st[3]);
  finalize_stats_kernel<<<1, 256, 0, stream>>>(st[3], g[3], bt[3], ss[3], 64, invK);
  maxk_kernel<<<(BN * 64) / 256, 256, 0, stream>>>(y2h, ss[3], x2);

  // ---- stage 3 ----
  rowsq64h_kernel<<<BN / 256, 256, 0, stream>>>(x2, sq);
  knn64_kernel<<<BB * (NN / 16), 256, knn_lds, stream>>>(x2, sq, idx);
  conv_edge64_kernel<<<Mk / 32, 256, 0, stream>>>(x2, idx, w5h, y1h, st[4]);
  finalize_stats_kernel<<<1, 256, 0, stream>>>(st[4], g[4], bt[4], ss[4], 64, invK);
  maxk_kernel<<<(BN * 64) / 256, 256, 0, stream>>>(y1h, ss[4], x3);

  // ---- final projection + output ----
  final_gemm_kernel<<<BN / 32, 256, 0, stream>>>(x1, x2, x3, w6h, yph, st[5]);
  finalize_stats_kernel<<<(1024 + 255) / 256, 256, 0, stream>>>(st[5], g[5], bt[5], ss[5], 1024, invBN);

  float* yout = (float*)d_out;  // (B,1024,N)
  int*   cidx = (int*)(yout + (size_t)BB * 1024 * NN);
  float* pool = (float*)(cidx + BB * 1024);
  final_out_kernel<<<BB * 16, 64, 0, stream>>>(yph, ss[5], yout, cidx, pool);
}
